// GRUD_cell_66340064854705
// MI455X (gfx1250) — matrix operand order repaired and checked by reading, not compiled, not run on hardware
//
#include <hip/hip_runtime.h>
#include <hip/hip_bf16.h>

typedef __attribute__((ext_vector_type(16))) _Float16 v16h;
typedef __attribute__((ext_vector_type(8)))  _Float16 v8h;
typedef __attribute__((ext_vector_type(8)))  float    v8f;

// Problem constants
constexpr int Bc = 256, Cc = 64, Hc = 128, Oc = 64, Tc = 512;
constexpr int BT  = 16;    // batch rows per workgroup
constexpr int KZR = 256;   // K of fused gate GEMMs: [xin(64) | hd(128) | m(64)]

// ---- LDS layout (units: halfs, then floats) ----
// Weight blocks are N-MAJOR (one K-row per output column): a WMMA B-fragment is
// 32 contiguous, 32B-aligned bytes per lane -> exactly 2x ds_load_b128.
constexpr int OFF_WZR  = 0;                     // [n=0..255][k=0..255] (z cols 0..127, r 128..255)
constexpr int OFF_WH   = OFF_WZR + 256 * 256;   // [n=0..127][k=0..255]
constexpr int OFF_WDG  = OFF_WH  + 128 * 256;   // [n=0..191][k=0..63]  ([gamma_x | gamma_h])
constexpr int OFF_WHY  = OFF_WDG + 192 * 64;    // [n=0..63][k=0..127]
constexpr int OFF_A    = OFF_WHY + 64 * 128;    // 16 x 256   gate GEMM A operand (row-major)
constexpr int OFF_D    = OFF_A   + 16 * 256;    // 16 x 64    delta tile (A for gamma)
constexpr int OFF_HN16 = OFF_D   + 16 * 64;     // 16 x 128   h_new f16 (A for y)
constexpr int HALF_TOT = OFF_HN16 + 16 * 128;

constexpr int FOFF_HF  = 0;                     // 16 x 128  h (f32, carried)
constexpr int FOFF_HD  = FOFF_HF + 16 * 128;    // 16 x 128  hd = gamma_h * h
constexpr int FOFF_XL  = FOFF_HD + 16 * 128;    // 16 x 64   x_last (carried)
constexpr int FOFF_XM  = FOFF_XL + 16 * 64;     // 16 x 64   x_mean
constexpr int FOFF_GX  = FOFF_XM + 16 * 64;     // 16 x 64   gamma_x
constexpr int FOFF_GH  = FOFF_GX + 16 * 64;     // 16 x 128  gamma_h
constexpr int FOFF_MF  = FOFF_GH + 16 * 128;    // 16 x 64   m (f32)
constexpr int FOFF_XF  = FOFF_MF + 16 * 64;     // 16 x 64   x (f32)
constexpr int FOFF_BDG = FOFF_XF + 16 * 64;     // 192
constexpr int FOFF_BMZ = FOFF_BDG + 192;        // 128
constexpr int FOFF_BMH = FOFF_BMZ + 128;        // 128
constexpr int FOFF_BHY = FOFF_BMH + 128;        // 64
constexpr int FLOAT_TOT = FOFF_BHY + 64;

constexpr size_t SMEM_BYTES = (size_t)HALF_TOT * 2 + (size_t)FLOAT_TOT * 4; // ~292 KB < 320 KB

__device__ __forceinline__ float sigmoidf_(float x) { return 1.f / (1.f + __expf(-x)); }

// ---- vectorized fragment loads -------------------------------------------
// A fragment (16-bit A 16x32): lane holds K = [kb..kb+15] (contiguous,
// matching B) -> two aligned 16B chunks.
__device__ __forceinline__ v16h ld_frag_a(const _Float16* p) {
  v8h lo = *(const v8h*)(p);
  v8h hi = *(const v8h*)(p + 8);
  return __builtin_shufflevector(lo, hi, 0, 1, 2, 3, 4, 5, 6, 7,
                                 8, 9, 10, 11, 12, 13, 14, 15);
}
// B fragment: 16 contiguous halfs (N-major weight storage), 32B aligned.
__device__ __forceinline__ v16h ld_frag_b(const _Float16* p) {
  return *(const v16h*)(p);
}

// ---- WMMA tile helpers ----------------------------------------------------
template <int K, int LDA, int LDK>
__device__ __forceinline__ v8f gemm_tile(const _Float16* __restrict__ A,
                                         const _Float16* __restrict__ Bm,
                                         int n0, int lane) {
  v8f acc = {};
  const int row = lane & 15;
  const int akb = (lane & 16) ? 16 : 0;
  const int col = lane & 15;
  const int bkb = (lane & 16) ? 16 : 0;
  const _Float16* arow = A + row * LDA + akb;
  const _Float16* brow = Bm + (n0 + col) * LDK + bkb;
#pragma unroll
  for (int k = 0; k < K; k += 32) {
    v16h a = ld_frag_a(arow + k);
    v16h b = ld_frag_b(brow + k);
    acc = __builtin_amdgcn_wmma_f32_16x16x32_f16(false, a, false, b, (short)0, acc,
                                                 false, false);
  }
  return acc;
}

// Two output tiles sharing one A operand: halves A-side DS traffic and gives
// the WMMA pipe two independent dependency chains.
template <int K, int LDA, int LDK>
__device__ __forceinline__ void gemm_tile2(const _Float16* __restrict__ A,
                                           const _Float16* __restrict__ Bm,
                                           int n0a, int n0b, int lane,
                                           v8f& acc0, v8f& acc1) {
  const int row = lane & 15;
  const int akb = (lane & 16) ? 16 : 0;
  const int col = lane & 15;
  const int bkb = (lane & 16) ? 16 : 0;
  const _Float16* arow  = A + row * LDA + akb;
  const _Float16* b0row = Bm + (n0a + col) * LDK + bkb;
  const _Float16* b1row = Bm + (n0b + col) * LDK + bkb;
#pragma unroll
  for (int k = 0; k < K; k += 32) {
    v16h a  = ld_frag_a(arow + k);
    v16h b0 = ld_frag_b(b0row + k);
    v16h b1 = ld_frag_b(b1row + k);
    acc0 = __builtin_amdgcn_wmma_f32_16x16x32_f16(false, a, false, b0, (short)0, acc0,
                                                  false, false);
    acc1 = __builtin_amdgcn_wmma_f32_16x16x32_f16(false, a, false, b1, (short)0, acc1,
                                                  false, false);
  }
}

__global__ __launch_bounds__(256, 1)
void grud_fused_kernel(const float* __restrict__ input,  const float* __restrict__ x_mean,
                       const float* __restrict__ lastx,
                       const float* __restrict__ w_dg_x, const float* __restrict__ b_dg_x,
                       const float* __restrict__ w_dg_h, const float* __restrict__ b_dg_h,
                       const float* __restrict__ w_xz,   const float* __restrict__ w_hz,
                       const float* __restrict__ w_mz,   const float* __restrict__ b_mz,
                       const float* __restrict__ w_xr,   const float* __restrict__ w_hr,
                       const float* __restrict__ w_mr,
                       const float* __restrict__ w_xh,   const float* __restrict__ w_hh,
                       const float* __restrict__ w_mh,   const float* __restrict__ b_mh,
                       const float* __restrict__ w_hy,   const float* __restrict__ b_hy,
                       float* __restrict__ ys_out, float* __restrict__ hs_out) {
  extern __shared__ char smem_raw[];
  _Float16* sh   = (_Float16*)smem_raw;
  _Float16* sWzr = sh + OFF_WZR;
  _Float16* sWh  = sh + OFF_WH;
  _Float16* sWdg = sh + OFF_WDG;
  _Float16* sWhy = sh + OFF_WHY;
  _Float16* sA   = sh + OFF_A;
  _Float16* sD   = sh + OFF_D;
  _Float16* sHn16= sh + OFF_HN16;
  float* sf  = (float*)(sh + HALF_TOT);
  float* sHf = sf + FOFF_HF;  float* sHd = sf + FOFF_HD;
  float* sXl = sf + FOFF_XL;  float* sXm = sf + FOFF_XM;
  float* sGx = sf + FOFF_GX;  float* sGh = sf + FOFF_GH;
  float* sMf = sf + FOFF_MF;  float* sXf = sf + FOFF_XF;
  float* sBdg= sf + FOFF_BDG; float* sBmz= sf + FOFF_BMZ;
  float* sBmh= sf + FOFF_BMH; float* sBhy= sf + FOFF_BHY;

  const int tid  = threadIdx.x;
  const int lane = tid & 31;
  // Wave id as a compiler-provable scalar -> wave-uniform branches become
  // s_cbranch (no exec-mask save/restore around WMMA regions).
  const int wave = __builtin_amdgcn_readfirstlane(tid >> 5);   // 8 waves (wave32)
  const int b0   = blockIdx.x * BT;
  const int colc = lane & 15;                 // C/D tile column
  const int rbase = (lane & 16) ? 8 : 0;      // C/D tile row base

  // ---------------- one-time staging: weights -> f16 N-major LDS ----------------
  for (int idx = tid; idx < 256 * 256; idx += 256) {   // sWzr[n*256 + k]
    int n = idx >> 8, k = idx & 255, nn = n & 127;
    float v;
    if (n < 128)
      v = (k < 64) ? w_xz[k * Hc + nn] : (k < 192) ? w_hz[(k - 64) * Hc + nn]
                                                   : w_mz[(k - 192) * Hc + nn];
    else
      v = (k < 64) ? w_xr[k * Hc + nn] : (k < 192) ? w_hr[(k - 64) * Hc + nn]
                                                   : w_mr[(k - 192) * Hc + nn];
    sWzr[idx] = (_Float16)v;
  }
  for (int idx = tid; idx < 128 * 256; idx += 256) {   // sWh[n*256 + k]
    int n = idx >> 8, k = idx & 255;
    float v = (k < 64) ? w_xh[k * Hc + n] : (k < 192) ? w_hh[(k - 64) * Hc + n]
                                                      : w_mh[(k - 192) * Hc + n];
    sWh[idx] = (_Float16)v;
  }
  for (int idx = tid; idx < 192 * 64; idx += 256) {    // sWdg[n*64 + k]
    int n = idx >> 6, k = idx & 63;
    float v = (n < 64) ? w_dg_x[k * Cc + n] : w_dg_h[k * Hc + (n - 64)];
    sWdg[idx] = (_Float16)v;
  }
  for (int idx = tid; idx < 64 * 128; idx += 256) {    // sWhy[n*128 + k]
    int n = idx >> 7, k = idx & 127;
    sWhy[idx] = (_Float16)w_hy[k * Oc + n];
  }

  for (int idx = tid; idx < BT * Cc; idx += 256) {
    int r = idx >> 6, c = idx & 63;
    sXm[idx] = x_mean[(size_t)(b0 + r) * Cc + c];
    sXl[idx] = lastx[(size_t)(b0 + r) * Cc + c];
  }
  for (int idx = tid; idx < BT * Hc; idx += 256) sHf[idx] = 0.f;  // h0 = 0
  if (tid < 192) sBdg[tid] = (tid < 64) ? b_dg_x[tid] : b_dg_h[tid - 64];
  if (tid < 128) { sBmz[tid] = b_mz[tid]; sBmh[tid] = b_mh[tid]; }
  if (tid < 64)  sBhy[tid] = b_hy[tid];
  __syncthreads();

  // ---------------- sequential scan over T ----------------
  for (int t = 0; t < Tc; ++t) {
    // stage 1: load x/m/d tiles; prefetch one cacheline ahead along t
    for (int idx = tid; idx < BT * Cc; idx += 256) {
      int r = idx >> 6, c = idx & 63;
      size_t rowb = (size_t)(b0 + r) * 3 * Cc;
      const float* px = &input[(rowb + 0 * Cc + c) * Tc + t];
      const float* pm = &input[(rowb + 1 * Cc + c) * Tc + t];
      const float* pd = &input[(rowb + 2 * Cc + c) * Tc + t];
      if (t + 16 < Tc) {
        __builtin_prefetch(px + 16, 0, 0);
        __builtin_prefetch(pm + 16, 0, 0);
        __builtin_prefetch(pd + 16, 0, 0);
      }
      float x = *px, m = *pm, d = *pd;
      sXf[idx] = x;
      sMf[idx] = m;
      sD[idx]  = (_Float16)d;
      sA[r * KZR + 192 + c] = (_Float16)m;   // m-section of fused gate operand
    }
    __syncthreads();

    // stage 2: gamma = exp(-relu(d @ [Wdgx|Wdgh] + b)); 12 tiles, scalar branching
    if (wave < 4) {                    // tiles w (-> gamma_x) and w+8 (-> gamma_h)
      v8f ga = {}, gb = {};
      const int n0a = wave * 16, n0b = (wave + 8) * 16;
      gemm_tile2<64, 64, 64>(sD, sWdg, n0a, n0b, lane, ga, gb);
#pragma unroll
      for (int i = 0; i < 8; ++i) {
        int row = rbase + i;
        sGx[row * 64  + n0a + colc]      = __expf(-fmaxf(ga[i] + sBdg[n0a + colc], 0.f));
        sGh[row * 128 + n0b + colc - 64] = __expf(-fmaxf(gb[i] + sBdg[n0b + colc], 0.f));
      }
    } else {                           // tiles 4..7 (-> gamma_h)
      const int n0g = wave * 16;
      v8f ga = gemm_tile<64, 64, 64>(sD, sWdg, n0g, lane);
#pragma unroll
      for (int i = 0; i < 8; ++i) {
        int row = rbase + i, n = n0g + colc;
        sGh[row * 128 + n - 64] = __expf(-fmaxf(ga[i] + sBdg[n], 0.f));
      }
    }
    __syncthreads();

    // stage 3: x_last / xin / hd elementwise; build A = [xin | hd | m] (f16)
    for (int idx = tid; idx < BT * Cc; idx += 256) {
      int r = idx >> 6, c = idx & 63;
      float m = sMf[idx], x = sXf[idx];
      float xl = (m > 0.f) ? x : sXl[idx];
      sXl[idx] = xl;
      float gx = sGx[idx];
      float xin = m * x + (1.f - m) * (gx * xl + (1.f - gx) * sXm[idx]);
      sA[r * KZR + c] = (_Float16)xin;
    }
    for (int idx = tid; idx < BT * Hc; idx += 256) {
      int r = idx >> 7, c = idx & 127;
      float hd = sGh[idx] * sHf[idx];
      sHd[idx] = hd;
      sA[r * KZR + 64 + c] = (_Float16)hd;
    }
    __syncthreads();

    // stage 4: z and r gates in one pass over A (shared A fragments), K=256
    const int n0 = wave * 16;
    v8f zacc = {}, racc = {};
    gemm_tile2<KZR, KZR, 256>(sA, sWzr, n0, 128 + n0, lane, zacc, racc);
    float zv[8], rv[8];
#pragma unroll
    for (int i = 0; i < 8; ++i) {
      zv[i] = sigmoidf_(zacc[i] + sBmz[n0 + colc]);
      rv[i] = sigmoidf_(racc[i]);
    }
    __syncthreads();                       // all waves done reading sA[hd] before overwrite
#pragma unroll
    for (int i = 0; i < 8; ++i) {          // overwrite hd-section with r*hd
      int row = rbase + i, n = n0 + colc;
      sA[row * KZR + 64 + n] = (_Float16)(rv[i] * sHd[row * Hc + n]);
    }
    __syncthreads();

    // stage 5: h_tilde = tanh([xin | r*hd | m] @ Wh + b); h_new; stream hs out
    v8f hacc = gemm_tile<KZR, KZR, 256>(sA, sWh, n0, lane);
#pragma unroll
    for (int i = 0; i < 8; ++i) {
      int row = rbase + i, n = n0 + colc;
      float ht = tanhf(hacc[i] + sBmh[n]);
      float hd = sHd[row * Hc + n];
      float hn = (1.f - zv[i]) * hd + zv[i] * ht;
      sHf[row * Hc + n]   = hn;
      sHn16[row * Hc + n] = (_Float16)hn;
      hs_out[((size_t)(b0 + row) * Tc + t) * Hc + n] = hn;
    }
    __syncthreads();

    // stage 6: y = sigmoid(h_new @ Why + b); N=64 -> waves 0..3
    if (wave < 4) {
      v8f yacc = gemm_tile<128, 128, 128>(sHn16, sWhy, n0, lane);
#pragma unroll
      for (int i = 0; i < 8; ++i) {
        int row = rbase + i, n = n0 + colc;
        ys_out[((size_t)(b0 + row) * Tc + t) * Oc + n] = sigmoidf_(yacc[i] + sBhy[n]);
      }
    }
    __syncthreads();
  }
}

extern "C" void kernel_launch(void* const* d_in, const int* in_sizes, int n_in,
                              void* d_out, int out_size, void* d_ws, size_t ws_size,
                              hipStream_t stream) {
  (void)in_sizes; (void)n_in; (void)out_size; (void)d_ws; (void)ws_size;
  const float* input  = (const float*)d_in[0];
  const float* x_mean = (const float*)d_in[1];
  const float* lastx  = (const float*)d_in[2];
  const float* w_dg_x = (const float*)d_in[3];
  const float* b_dg_x = (const float*)d_in[4];
  const float* w_dg_h = (const float*)d_in[5];
  const float* b_dg_h = (const float*)d_in[6];
  const float* w_xz   = (const float*)d_in[7];
  const float* w_hz   = (const float*)d_in[8];
  const float* w_mz   = (const float*)d_in[9];
  const float* b_mz   = (const float*)d_in[10];
  const float* w_xr   = (const float*)d_in[11];
  const float* w_hr   = (const float*)d_in[12];
  const float* w_mr   = (const float*)d_in[13];
  const float* w_xh   = (const float*)d_in[14];
  const float* w_hh   = (const float*)d_in[15];
  const float* w_mh   = (const float*)d_in[16];
  const float* b_mh   = (const float*)d_in[17];
  const float* w_hy   = (const float*)d_in[18];
  const float* b_hy   = (const float*)d_in[19];

  float* ys = (float*)d_out;                         // (B, T, O)
  float* hs = ys + (size_t)Bc * Tc * Oc;             // (B, T, H)

  dim3 grid(Bc / BT), block(256);
  hipLaunchKernelGGL(grud_fused_kernel, grid, block, SMEM_BYTES, stream,
                     input, x_mean, lastx,
                     w_dg_x, b_dg_x, w_dg_h, b_dg_h,
                     w_xz, w_hz, w_mz, b_mz,
                     w_xr, w_hr, w_mr,
                     w_xh, w_hh, w_mh, b_mh,
                     w_hy, b_hy, ys, hs);
}
